// FRAP_56410100465871
// MI455X (gfx1250) — matrix operand order repaired and checked by reading, not compiled, not run on hardware
//
#include <hip/hip_runtime.h>

typedef __attribute__((ext_vector_type(16))) _Float16 v16h;
typedef __attribute__((ext_vector_type(8)))  float    v8f;

#define WS_YW    0      // 1280 floats: yw[8][8][20]
#define WS_CB    1280   // 1 float: cbias
#define WS_B     1296   // 1024 dwords: 4 packed B tiles (32 lanes x 8 dwords each)

__device__ __forceinline__ float relu(float a) { return fmaxf(a, 0.0f); }

// Within-wave LDS producer->consumer ordering: LDS ops from one wave complete
// in order (DScnt, in-order per ISA); just stop compiler reordering + drain.
__device__ __forceinline__ void wave_sync() {
  __builtin_amdgcn_wave_barrier();
  asm volatile("s_wait_dscnt 0" ::: "memory");
  __builtin_amdgcn_wave_barrier();
}

// ---------------------------------------------------------------------------
// Precompute kernel:
//   yw[i][j][k] = y(mask[i,j])[k] * (wk1@wk2)[k]        (8*8*20 floats)
//   cbias       = bk1@wk2 + bk2
//   packed WMMA B tiles for wc1 (K=32) and wc2 (K=20 pad 32), both N-tiles,
//   in the 16-bit 32x16 B layout: lane -> col n = lane%16; lanes 0-15 hold
//   K=0..7,16..23 (dword p -> K=2p+8*(p>>2), +1), lanes 16-31 hold K+8 of those.
// ---------------------------------------------------------------------------
__global__ void frap_pre(
    const float* __restrict__ mask,
    const float* __restrict__ wm1, const float* __restrict__ bm1,
    const float* __restrict__ wm2, const float* __restrict__ bm2,
    const float* __restrict__ wm3, const float* __restrict__ bm3,
    const float* __restrict__ wk1, const float* __restrict__ bk1,
    const float* __restrict__ wk2, const float* __restrict__ bk2,
    const float* __restrict__ wc1, const float* __restrict__ wc2,
    float* __restrict__ ws)
{
  const int t = threadIdx.x;
  if (t >= 64) return;

  float wk[20];
#pragma unroll
  for (int k = 0; k < 20; ++k) {
    float a = 0.0f;
#pragma unroll
    for (int c = 0; c < 8; ++c) a += wk1[k*8 + c] * wk2[c];
    wk[k] = a;
  }

  const float m = mask[t];
  float t1[4];
#pragma unroll
  for (int c = 0; c < 4; ++c) t1[c] = fmaxf(m * wm1[c] + bm1[c], 0.0f);

  float t2[20];
#pragma unroll
  for (int d = 0; d < 20; ++d) {
    float a = bm2[d];
#pragma unroll
    for (int c = 0; c < 4; ++c) a += t1[c] * wm2[c*20 + d];
    t2[d] = fmaxf(a, 0.0f);
  }

#pragma unroll
  for (int e = 0; e < 20; ++e) {
    float a = bm3[e];
#pragma unroll
    for (int d = 0; d < 20; ++d) a += t2[d] * wm3[d*20 + e];
    ws[WS_YW + t*20 + e] = fmaxf(a, 0.0f) * wk[e];
  }

  if (t == 0) {
    float cb = bk2[0];
#pragma unroll
    for (int c = 0; c < 8; ++c) cb += bk1[c] * wk2[c];
    ws[WS_CB] = cb;
  }

  // ---- pack B tiles: tile 0/1 = wc1 N-tile0/1, tile 2/3 = wc2 N-tile0/1 ----
  for (int s = t; s < 128; s += 64) {
    const int tile = s >> 5;          // 0..3
    const int lane = s & 31;
    const int grp  = lane >> 4;
    const int ln   = lane & 15;
    const int colB = (16 + ln < 20) ? (16 + ln) : 0;
    const float mB = (16 + ln < 20) ? 1.0f : 0.0f;
    const int   n  = (tile & 1) ? colB : ln;
    const float nm = (tile & 1) ? mB : 1.0f;
    const float* W = (tile < 2) ? wc1 : wc2;
    const int Kmax = (tile < 2) ? 32 : 20;
    unsigned* dst = (unsigned*)(ws + WS_B) + s*8;
    for (int p = 0; p < 8; ++p) {
      const int k0 = 2*p + 8*grp + ((p >> 2) << 3), k1 = k0 + 1;
      float f0 = 0.0f, f1 = 0.0f;
      if (k0 < Kmax) f0 = W[k0*20 + n] * nm;
      if (k1 < Kmax) f1 = W[k1*20 + n] * nm;
      union { _Float16 h[2]; unsigned u; } cv;
      cv.h[0] = (_Float16)f0;
      cv.h[1] = (_Float16)f1;
      dst[p] = cv.u;
    }
  }
}

// ---------------------------------------------------------------------------
// Main kernel: 8 waves / block, one batch element per wave per iteration.
// GEMM steps (pair@wc1 and h1@wc2) run on v_wmma_f32_16x16x32_f16.
// s_pd/s_u/s_v rows are padded to 32 floats with a permanently-zero tail so
// WMMA A-matrix zero-halves come from address selection, not lane masking.
// ---------------------------------------------------------------------------
__global__ __launch_bounds__(256) void frap_main(
    const float* __restrict__ x,
    const float* __restrict__ wv1, const float* __restrict__ bv1,
    const float* __restrict__ wv2, const float* __restrict__ bv2,
    const float* __restrict__ wp1, const float* __restrict__ bp1,
    const float* __restrict__ wp2, const float* __restrict__ bp2,
    const float* __restrict__ we,  const float* __restrict__ be,
    const float* __restrict__ bc1, const float* __restrict__ bc2,
    const float* __restrict__ ws,
    float* __restrict__ out, int nbatch)
{
  __shared__ float s_we[128];
  __shared__ float s_be[16];
  __shared__ float s_bc1[20];
  __shared__ float s_bc2[32];       // padded, cols 20..31 = 0
  __shared__ float s_yw[1280];
  __shared__ float s_small[32];     // wv1,bv1,wv2,bv2 | wp1,bp1,wp2,bp2
  __shared__ float s_cbias;
  __shared__ float s_d8 [8][64];
  __shared__ float s_dem[8][128];
  __shared__ float s_pd [8][256];   // 8 rows x 32 (cols 16..31 always 0)
  __shared__ float s_u  [8][256];   // 8 rows x 32 (cols 20..31 always 0)
  __shared__ float s_v  [8][256];   // 8 rows x 32 (cols 20..31 always 0)

  const int tid  = threadIdx.x;
  const int lane = tid & 31;
  const int wv   = tid >> 5;
  const int grp  = lane >> 4;       // 0 or 1 (16-lane half)
  const int ln16 = lane & 15;

  for (int i = tid; i < 128;  i += 256) s_we[i]  = we[i];
  for (int i = tid; i < 16;   i += 256) s_be[i]  = be[i];
  for (int i = tid; i < 20;   i += 256) s_bc1[i] = bc1[i];
  for (int i = tid; i < 32;   i += 256) s_bc2[i] = (i < 20) ? bc2[i] : 0.0f;
  for (int i = tid; i < 1280; i += 256) s_yw[i]  = ws[WS_YW + i];
  for (int i = tid; i < 2048; i += 256) {       // zero padded scratch (tails stay 0)
    ((float*)s_pd)[i] = 0.0f;
    ((float*)s_u)[i]  = 0.0f;
    ((float*)s_v)[i]  = 0.0f;
  }
  if (tid == 0) {
#pragma unroll
    for (int i = 0; i < 2; ++i) s_small[i]      = wv1[i];
#pragma unroll
    for (int i = 0; i < 2; ++i) s_small[2 + i]  = bv1[i];
#pragma unroll
    for (int i = 0; i < 8; ++i) s_small[4 + i]  = wv2[i];
#pragma unroll
    for (int i = 0; i < 4; ++i) s_small[12 + i] = bv2[i];
#pragma unroll
    for (int i = 0; i < 2; ++i) s_small[16 + i] = wp1[i];
#pragma unroll
    for (int i = 0; i < 2; ++i) s_small[18 + i] = bp1[i];
#pragma unroll
    for (int i = 0; i < 8; ++i) s_small[20 + i] = wp2[i];
#pragma unroll
    for (int i = 0; i < 4; ++i) s_small[28 + i] = bp2[i];
    s_cbias = ws[WS_CB];
  }
  __syncthreads();

  const int  col1c = (16 + ln16 < 20) ? (16 + ln16) : 0;  // clamped col, N-tile 1
  const float yw1m = (16 + ln16 < 20) ? 1.0f : 0.0f;

  // ---- load pre-packed WMMA B tiles (f16) from workspace ----
  v16h bw0, bw1, bt0, bt1;   // wc1 N0/N1, wc2 N0/N1
  {
    const uint4* base = (const uint4*)(ws + WS_B);
    union { uint4 q[2]; v16h v; } L;
    L.q[0] = base[(0*32 + lane)*2 + 0]; L.q[1] = base[(0*32 + lane)*2 + 1]; bw0 = L.v;
    L.q[0] = base[(1*32 + lane)*2 + 0]; L.q[1] = base[(1*32 + lane)*2 + 1]; bw1 = L.v;
    L.q[0] = base[(2*32 + lane)*2 + 0]; L.q[1] = base[(2*32 + lane)*2 + 1]; bt0 = L.v;
    L.q[0] = base[(3*32 + lane)*2 + 0]; L.q[1] = base[(3*32 + lane)*2 + 1]; bt1 = L.v;
  }
  // ---- bc2 as C matrix for the h2 GEMM (D: VGPR r -> row r+8*grp, col lane%16)
  v8f c0, c1;
  {
    const float b0v = s_bc2[ln16];
    const float b1v = s_bc2[16 + ln16];
#pragma unroll
    for (int r = 0; r < 8; ++r) { c0[r] = b0v; c1[r] = b1v; }
  }
  // ---- bc1 as C matrix for the u/v GEMM: u-rows (grp0 lanes) get bc1, v-rows 0
  v8f cu0, cu1;
  {
    const float bcv0 = grp ? 0.0f : s_bc1[ln16];
    const float bcv1 = grp ? 0.0f : (s_bc1[col1c] * yw1m);
#pragma unroll
    for (int r = 0; r < 8; ++r) { cu0[r] = bcv0; cu1[r] = bcv1; }
  }

  // store row permutation for the folded reduction:
  // lane l ends holding out[4*l0 + 2*l1 + 1*l2]
  const int sigma = ((ln16 & 1) << 2) | (ln16 & 2) | ((ln16 >> 2) & 1);

  const int waveGlobal = blockIdx.x * 8 + wv;
  const int nWaves = gridDim.x * 8;

  for (int b = waveGlobal; b < nbatch; b += nWaves) {
    { const int bn = b + nWaves; if (bn < nbatch) __builtin_prefetch(&x[bn*16], 0, 3); }

    // ---- step 1: per-scalar tiny MLPs -> d8 (8 rows x 8) ----
    if (lane < 16) {
      const float s  = x[(size_t)b*16 + lane];
      const int off  = (lane < 8) ? 0 : 16;
      const float t0 = relu(s * s_small[off+0] + s_small[off+2]);
      const float t1 = relu(s * s_small[off+1] + s_small[off+3]);
      const int row  = lane & 7;
      const int cb   = (lane < 8) ? 0 : 4;
#pragma unroll
      for (int c = 0; c < 4; ++c) {
        const float d = relu(t0 * s_small[off+4+c] + t1 * s_small[off+8+c] + s_small[off+12+c]);
        s_d8[wv][row*8 + cb + c] = d;
      }
    }
    wave_sync();

    // ---- step 2: demand = relu(d8 @ we + be)  (8x16) ----
    {
      const int i  = lane >> 2;
      const int nb = (lane & 3) * 4;
      float a0 = s_be[nb+0], a1 = s_be[nb+1], a2 = s_be[nb+2], a3 = s_be[nb+3];
#pragma unroll
      for (int k = 0; k < 8; ++k) {
        const float dk = s_d8[wv][i*8 + k];
        a0 += dk * s_we[k*16 + nb + 0];
        a1 += dk * s_we[k*16 + nb + 1];
        a2 += dk * s_we[k*16 + nb + 2];
        a3 += dk * s_we[k*16 + nb + 3];
      }
      s_dem[wv][i*16 + nb + 0] = relu(a0);
      s_dem[wv][i*16 + nb + 1] = relu(a1);
      s_dem[wv][i*16 + nb + 2] = relu(a2);
      s_dem[wv][i*16 + nb + 3] = relu(a3);
    }
    wave_sync();

    // ---- step 3: pd[i] = demand[PA[i]] + demand[PB[i]]  (8x16, stride 32) ----
#pragma unroll
    for (int q = 0; q < 4; ++q) {
      const int idx = lane*4 + q;
      const int i = idx >> 4, n = idx & 15;
      const int pa = (int)((0x36221400u >> (4*i)) & 0xFu);  // {0,0,4,1,2,2,6,3}
      const int pb = (int)((0x77365514u >> (4*i)) & 0xFu);  // {4,1,5,5,6,3,7,7}
      s_pd[wv][i*32 + n] = s_dem[wv][pa*16 + n] + s_dem[wv][pb*16 + n];
    }
    wave_sync();

    // ---- step 4 (WMMA): one 16x32 A packs both halves of wc1:
    //        rows 0..7  = [pd_i | 0 ]  -> u = pd@wc1[0:16] + bc1
    //        rows 8..15 = [ 0 | pd_i]  -> v = pd@wc1[16:32]
    //      Zero halves come from the zero-padded pd tail via address select. ----
    {
      const int  m   = ln16;
      const bool isU = (m < 8);
      const int  src = isU ? m : (m - 8);
      const int offA = 8*grp + (isU ? 0 : 16);   // elems 0..7  (K 0..15 half)
      const int offB = 8*grp + (isU ? 16 : 0);   // elems 8..15 (K 16..31 half)
      const float4* p0 = (const float4*)&s_pd[wv][src*32 + offA];
      const float4* p1 = (const float4*)&s_pd[wv][src*32 + offB];
      const float4 a0 = p0[0], a1 = p0[1];
      const float4 b0 = p1[0], b1 = p1[1];
      v16h A;
      A[0]  = (_Float16)a0.x; A[1]  = (_Float16)a0.y;
      A[2]  = (_Float16)a0.z; A[3]  = (_Float16)a0.w;
      A[4]  = (_Float16)a1.x; A[5]  = (_Float16)a1.y;
      A[6]  = (_Float16)a1.z; A[7]  = (_Float16)a1.w;
      A[8]  = (_Float16)b0.x; A[9]  = (_Float16)b0.y;
      A[10] = (_Float16)b0.z; A[11] = (_Float16)b0.w;
      A[12] = (_Float16)b1.x; A[13] = (_Float16)b1.y;
      A[14] = (_Float16)b1.z; A[15] = (_Float16)b1.w;

      v8f U0 = __builtin_amdgcn_wmma_f32_16x16x32_f16(false, A, false, bw0,
                                                      (short)0, cu0, false, false);
      v8f U1 = __builtin_amdgcn_wmma_f32_16x16x32_f16(false, A, false, bw1,
                                                      (short)0, cu1, false, false);
      // scatter: grp0 lanes hold u rows 0..7, grp1 lanes hold v rows 0..7
      float* dst = grp ? &s_v[wv][0] : &s_u[wv][0];
#pragma unroll
      for (int r = 0; r < 8; ++r) dst[r*32 + ln16] = U0[r];
      if (16 + ln16 < 20) {
#pragma unroll
        for (int r = 0; r < 8; ++r) dst[r*32 + 16 + ln16] = U1[r];
      }
    }
    wave_sync();

    // ---- step 5: 4 tiles of 16 rows: h2 = relu(h1@wc2 + bc2) via WMMA,
    //              then out = <relu(h2), yw> + cbias ----
#pragma unroll 2
    for (int t = 0; t < 4; ++t) {
      const int i = 2*t + (ln16 >> 3);
      const int j = ln16 & 7;
      // A-half K offsets per lane group: grp0 -> K0..7 & K16..23,
      //                                  grp1 -> K8..15 & K24..31 (zeros in tail)
      const float* up = &s_u[wv][i*32 + 8*grp];
      const float* vp = &s_v[wv][j*32 + 8*grp];
      const float4 u0 = *(const float4*)(up +  0);
      const float4 u1 = *(const float4*)(up +  4);
      const float4 u2 = *(const float4*)(up + 16);
      const float4 u3 = *(const float4*)(up + 20);
      const float4 v0 = *(const float4*)(vp +  0);
      const float4 v1 = *(const float4*)(vp +  4);
      const float4 v2 = *(const float4*)(vp + 16);
      const float4 v3 = *(const float4*)(vp + 20);
      v16h A;
      A[0]  = (_Float16)relu(u0.x + v0.x); A[1]  = (_Float16)relu(u0.y + v0.y);
      A[2]  = (_Float16)relu(u0.z + v0.z); A[3]  = (_Float16)relu(u0.w + v0.w);
      A[4]  = (_Float16)relu(u1.x + v1.x); A[5]  = (_Float16)relu(u1.y + v1.y);
      A[6]  = (_Float16)relu(u1.z + v1.z); A[7]  = (_Float16)relu(u1.w + v1.w);
      A[8]  = (_Float16)relu(u2.x + v2.x); A[9]  = (_Float16)relu(u2.y + v2.y);
      A[10] = (_Float16)relu(u2.z + v2.z); A[11] = (_Float16)relu(u2.w + v2.w);
      A[12] = (_Float16)relu(u3.x + v3.x); A[13] = (_Float16)relu(u3.y + v3.y);
      A[14] = (_Float16)relu(u3.z + v3.z); A[15] = (_Float16)relu(u3.w + v3.w);

      v8f D0 = __builtin_amdgcn_wmma_f32_16x16x32_f16(false, A, false, bt0,
                                                      (short)0, c0, false, false);
      v8f D1 = __builtin_amdgcn_wmma_f32_16x16x32_f16(false, A, false, bt1,
                                                      (short)0, c1, false, false);

      // D layout: VGPR r, lane -> row r + 8*grp, col lane%16.
      // Row m = r + 8*grp of tile t  ->  i = 2t+grp, j = r.
      const int it = 2*t + grp;
      float part[8];
#pragma unroll
      for (int r = 0; r < 8; ++r) {
        const float* ywrow = &s_yw[(it*8 + r)*20];
        const float h0 = relu(D0[r]);
        const float h1x = relu(D1[r]);
        part[r] = h0 * ywrow[ln16] + h1x * (yw1m * ywrow[col1c]);
      }

      // ---- folded butterfly: reduce 8 rows over 16 lanes with 8 shuffles.
      // Each stage halves the register count by lane-selecting which register
      // crosses the xor exchange; row r's sum lands on lanes with
      // (l0,l1,l2) = bits of r (sigma mapping precomputed).
      float q4[4];
#pragma unroll
      for (int r = 0; r < 4; ++r) {
        const float a = part[r], bb = part[r + 4];
        const float send = (ln16 & 1) ? a : bb;
        const float recv = __shfl_xor(send, 1, 32);
        q4[r] = ((ln16 & 1) ? bb : a) + recv;
      }
      float q2[2];
#pragma unroll
      for (int r = 0; r < 2; ++r) {
        const float a = q4[r], bb = q4[r + 2];
        const float send = (ln16 & 2) ? a : bb;
        const float recv = __shfl_xor(send, 2, 32);
        q2[r] = ((ln16 & 2) ? bb : a) + recv;
      }
      float q1;
      {
        const float a = q2[0], bb = q2[1];
        const float send = (ln16 & 4) ? a : bb;
        const float recv = __shfl_xor(send, 4, 32);
        q1 = ((ln16 & 4) ? bb : a) + recv;
      }
      q1 += __shfl_xor(q1, 8, 32);

      // lanes 0..7 of each 16-lane half store one output each
      if (ln16 < 8) {
        out[(size_t)b*64 + 16*t + 8*grp + sigma] = q1 + s_cbias;
      }
    }
  }
}

// ---------------------------------------------------------------------------
extern "C" void kernel_launch(void* const* d_in, const int* in_sizes, int n_in,
                              void* d_out, int out_size, void* d_ws, size_t ws_size,
                              hipStream_t stream) {
  (void)n_in; (void)out_size; (void)ws_size;
  const float* x    = (const float*)d_in[0];
  const float* mask = (const float*)d_in[1];
  const float* wv1  = (const float*)d_in[2];
  const float* bv1  = (const float*)d_in[3];
  const float* wv2  = (const float*)d_in[4];
  const float* bv2  = (const float*)d_in[5];
  const float* wp1  = (const float*)d_in[6];
  const float* bp1  = (const float*)d_in[7];
  const float* wp2  = (const float*)d_in[8];
  const float* bp2  = (const float*)d_in[9];
  const float* we   = (const float*)d_in[10];
  const float* be   = (const float*)d_in[11];
  const float* wc1  = (const float*)d_in[12];
  const float* bc1  = (const float*)d_in[13];
  const float* wc2  = (const float*)d_in[14];
  const float* bc2  = (const float*)d_in[15];
  const float* wm1  = (const float*)d_in[16];
  const float* bm1  = (const float*)d_in[17];
  const float* wm2  = (const float*)d_in[18];
  const float* bm2  = (const float*)d_in[19];
  const float* wm3  = (const float*)d_in[20];
  const float* bm3  = (const float*)d_in[21];
  const float* wk1  = (const float*)d_in[22];
  const float* bk1  = (const float*)d_in[23];
  const float* wk2  = (const float*)d_in[24];
  const float* bk2  = (const float*)d_in[25];

  float* ws  = (float*)d_ws;
  float* out = (float*)d_out;
  const int B = in_sizes[0] / 16;

  frap_pre<<<1, 64, 0, stream>>>(mask, wm1, bm1, wm2, bm2, wm3, bm3,
                                 wk1, bk1, wk2, bk2, wc1, wc2, ws);

  int blocks = 512;                       // 4096 waves -> 8 batches per wave
  if (blocks * 8 > B) blocks = (B + 7) / 8;
  if (blocks < 1) blocks = 1;
  frap_main<<<blocks, 256, 0, stream>>>(x, wv1, bv1, wv2, bv2, wp1, bp1, wp2, bp2,
                                        we, be, bc1, bc2, ws, out, B);
}
